// Attention_44332652429583
// MI455X (gfx1250) — matrix operand order repaired and checked by reading, not compiled, not run on hardware
//
#include <hip/hip_runtime.h>
#include <math.h>

#define BB   8
#define NN   8192
#define DQKC 128
#define DVC  128
#define MMC  256

typedef __attribute__((ext_vector_type(16))) __bf16 v16bf;
typedef __attribute__((ext_vector_type(8)))  __bf16 v8bf;
typedef __attribute__((ext_vector_type(8)))  float  v8f;

union AFragU { v16bf v; unsigned u[8]; };

// ---- monotonic float<->uint encoding for atomicMax on floats ----
__device__ __forceinline__ unsigned fenc(float f) {
  unsigned u = __float_as_uint(f);
  return (u & 0x80000000u) ? ~u : (u | 0x80000000u);
}
__device__ __forceinline__ float fdec(unsigned e) {
  unsigned u = (e & 0x80000000u) ? (e & 0x7fffffffu) : ~e;
  return __uint_as_float(u);
}

// 16-bit A-matrix 16x32 lane layout (contiguous k order: k = 16*(lane/16) + s)
__device__ __forceinline__ void a_layout(int lane, int* kidx, int& row) {
  row = lane & 15;
  int h16 = (lane >> 4) * 16;
#pragma unroll
  for (int v = 0; v < 8; ++v) {
    int kb = h16 + 2 * v;
    kidx[2 * v] = kb;
    kidx[2 * v + 1] = kb + 1;
  }
}

// Stage omega into LDS pre-swizzled into WMMA B-fragment layout:
// sw[(ks*16+mt)*32 + lane] = 16 contiguous bf16 for that lane's fragment.
__device__ __forceinline__ void stage_omega(const float* __restrict__ omega,
                                            v16bf* __restrict__ sw, int tid) {
  for (int p = tid; p < 64 * 32; p += 256) {
    int f = p >> 5, l = p & 31;
    int ks = f >> 4, mt = f & 15;
    int krow = ks * 32 + ((l >> 4) << 4);
    int col = mt * 16 + (l & 15);
    v16bf t;
#pragma unroll
    for (int i = 0; i < 16; ++i)
      t[i] = (__bf16)omega[(size_t)(krow + i) * MMC + col];
    sw[p] = t;
  }
}

// One wave computes a 16(n) x 256(M) strip of U = (X * d^-1/4) @ omega.
__device__ __forceinline__ void project_tiles(const float* __restrict__ X,
                                              const v16bf* __restrict__ sw,
                                              int b, int n_base, int lane,
                                              v8f accs[16]) {
  int row, kidx[16];
  a_layout(lane, kidx, row);
  const float scale = 0.29730177875068026f;  // 128^-0.25
  v16bf af[4];
#pragma unroll
  for (int ks = 0; ks < 4; ++ks) {
    const float2* xr2 =
        (const float2*)(X + ((size_t)b * NN + n_base + row) * DQKC + ks * 32);
#pragma unroll
    for (int v = 0; v < 8; ++v) {
      float2 p2 = xr2[kidx[2 * v] >> 1];
      af[ks][2 * v]     = (__bf16)(p2.x * scale);
      af[ks][2 * v + 1] = (__bf16)(p2.y * scale);
    }
  }
#pragma unroll
  for (int mt = 0; mt < 16; ++mt) {
    v8f acc = {0.f, 0.f, 0.f, 0.f, 0.f, 0.f, 0.f, 0.f};
#pragma unroll
    for (int ks = 0; ks < 4; ++ks) {
      v16bf bf = sw[(ks * 16 + mt) * 32 + lane];
      acc = __builtin_amdgcn_wmma_f32_16x16x32_bf16(false, af[ks], false, bf,
                                                    (short)0, acc, false, false);
    }
    accs[mt] = acc;
  }
}

__global__ __launch_bounds__(256) void init_kernel(unsigned* __restrict__ maxK) {
  int t = threadIdx.x;
  if (t < BB) maxK[t] = 0u;  // 0 encodes "below every float"
}

// Pass 1: per-batch max of U_K (before exp). atomicMax is order-independent.
__global__ __launch_bounds__(256) void kmax_kernel(const float* __restrict__ Kx,
                                                   const float* __restrict__ omega,
                                                   unsigned* __restrict__ maxK) {
  extern __shared__ char smem[];
  v16bf* sw = (v16bf*)smem;
  int tid = threadIdx.x;
  stage_omega(omega, sw, tid);
  __syncthreads();
  const int blkPerB = NN / 128;
  int b = blockIdx.x / blkPerB;
  int n_base = (blockIdx.x % blkPerB) * 128 + (tid >> 5) * 16;
  int lane = tid & 31;
  v8f accs[16];
  project_tiles(Kx, sw, b, n_base, lane, accs);
  float wmax = -3.0e38f;
#pragma unroll
  for (int mt = 0; mt < 16; ++mt)
#pragma unroll
    for (int r = 0; r < 8; ++r) wmax = fmaxf(wmax, accs[mt][r]);
#pragma unroll
  for (int off = 16; off >= 1; off >>= 1)
    wmax = fmaxf(wmax, __shfl_xor(wmax, off, 32));
  if (lane == 0) atomicMax(&maxK[b], fenc(wmax));
}

// Pass 2: phi. IS_Q: per-row max, store Qp[b][n][m].
//          !IS_Q: per-batch max, store KpT[b][m][n] (transposed, vector stores).
template <bool IS_Q>
__global__ __launch_bounds__(256) void phi_kernel(const float* __restrict__ X,
                                                  const float* __restrict__ omega,
                                                  const float* __restrict__ mask,
                                                  const unsigned* __restrict__ maxK,
                                                  __bf16* __restrict__ P) {
  extern __shared__ char smem[];
  v16bf* sw = (v16bf*)smem;                       // 64 KB
  float* hrow = (float*)(smem + 64 * 32 * 32);    // 128 f32
  float* mrow = hrow + 128;                       // 128 f32
  int tid = threadIdx.x;
  stage_omega(omega, sw, tid);
  const int blkPerB = NN / 128;
  int b = blockIdx.x / blkPerB;
  int n0 = (blockIdx.x % blkPerB) * 128;
  if (tid < 128) {
    const float* xr = X + ((size_t)b * NN + n0 + tid) * DQKC;
    float s = 0.f;
#pragma unroll 8
    for (int d = 0; d < DQKC; ++d) { float x = xr[d]; s += x * x; }
    hrow[tid] = s * 0.044194173824159216f;  // 1/(2*sqrt(128))
    mrow[tid] = mask[(size_t)b * NN + n0 + tid];
  }
  __syncthreads();

  int wave = tid >> 5, lane = tid & 31;
  int n_base = n0 + wave * 16;
  v8f accs[16];
  project_tiles(X, sw, b, n_base, lane, accs);

  int half = lane >> 4, col = lane & 15;
  float mx[8];
  if (IS_Q) {
#pragma unroll
    for (int r = 0; r < 8; ++r) {
      float v = -3.0e38f;
#pragma unroll
      for (int mt = 0; mt < 16; ++mt) v = fmaxf(v, accs[mt][r]);
#pragma unroll
      for (int off = 8; off >= 1; off >>= 1)
        v = fmaxf(v, __shfl_xor(v, off, 32));  // stays inside 16-lane half
      mx[r] = v;
    }
  } else {
    float v = fdec(maxK[b]);
#pragma unroll
    for (int r = 0; r < 8; ++r) mx[r] = v;
  }

  const float invSqrtM = 0.0625f;  // 1/sqrt(256)
#pragma unroll
  for (int mt = 0; mt < 16; ++mt) {
    if (IS_Q) {
#pragma unroll
      for (int r = 0; r < 8; ++r) {
        int lr = r + 8 * half;
        int n = n_base + lr;
        float val = (__expf(accs[mt][r] - hrow[wave * 16 + lr] - mx[r]) + 1e-4f) *
                    invSqrtM * mrow[wave * 16 + lr];
        P[((size_t)b * NN + n) * MMC + mt * 16 + col] = (__bf16)val;
      }
    } else {
      v8bf pk;
#pragma unroll
      for (int r = 0; r < 8; ++r) {
        int lr = r + 8 * half;
        float val = (__expf(accs[mt][r] - hrow[wave * 16 + lr] - mx[r]) + 1e-4f) *
                    invSqrtM * mrow[wave * 16 + lr];
        pk[r] = (__bf16)val;
      }
      int m = mt * 16 + col;
      *(v8bf*)(P + ((size_t)b * MMC + m) * NN + n_base + 8 * half) = pk;
    }
  }
}

// Ksum[b][m] = sum_n KpT[b][m][n]  (contiguous rows; deterministic)
__global__ __launch_bounds__(256) void ksum_kernel(const __bf16* __restrict__ KpT,
                                                   float* __restrict__ Ksum) {
  int wave = threadIdx.x >> 5, lane = threadIdx.x & 31;
  int gw = blockIdx.x * 8 + wave;  // [0, BB*MMC)
  const __bf16* r = KpT + (size_t)gw * NN;
  float s = 0.f;
#pragma unroll 4
  for (int i = 0; i < NN / 256; ++i) {
    v8bf c = *(const v8bf*)(r + (i * 32 + lane) * 8);
#pragma unroll
    for (int j = 0; j < 8; ++j) s += (float)c[j];
  }
#pragma unroll
  for (int off = 16; off >= 1; off >>= 1) s += __shfl_xor(s, off, 32);
  if (lane == 0) Ksum[gw] = s;
}

// KVpart[b][chunk][m][v] partial of KV over an n-chunk. Block = (b, chunk):
// 8 waves (one v-tile each) x 16 m-tiles; V and Kp read exactly once.
__global__ __launch_bounds__(256) void kv_kernel(const __bf16* __restrict__ KpT,
                                                 const float* __restrict__ V,
                                                 float* __restrict__ KVpart) {
  __shared__ v16bf sw_a[16 * 32];  // 16 KB: A-frags for all 16 m-tiles
  __shared__ v16bf sw_v[8 * 32];   //  8 KB: B-frags for all 8 v-tiles
  int tid = threadIdx.x;
  int wave = tid >> 5, lane = tid & 31;
  int b = blockIdx.x >> 4;
  int chunk = blockIdx.x & 15;
  int n_start = chunk * (NN / 16);

  v8f accs[16];
#pragma unroll
  for (int mt = 0; mt < 16; ++mt)
    accs[mt] = (v8f){0.f, 0.f, 0.f, 0.f, 0.f, 0.f, 0.f, 0.f};

  for (int it = 0; it < (NN / 16) / 32; ++it) {
    int n0 = n_start + it * 32;
    // stage A-fragments (Kp^T rows, contiguous dword pairs)
    for (int p = tid; p < 16 * 32; p += 256) {
      int mt = p >> 5, l = p & 31;
      int row, kidx[16];
      a_layout(l, kidx, row);
      const __bf16* ar = KpT + ((size_t)b * MMC + mt * 16 + row) * NN + n0;
      if (it + 1 < (NN / 16) / 32) __builtin_prefetch(ar + 32, 0, 0);
      AFragU af;
#pragma unroll
      for (int v = 0; v < 8; ++v)
        af.u[v] = *(const unsigned*)(ar + kidx[2 * v]);
      sw_a[p] = af.v;
    }
    // stage B-fragments (V rows, f32 -> bf16)
    {
      int p = tid;  // exactly 256 fragments
      int vt = p >> 5, l = p & 31;
      int col = vt * 16 + (l & 15);
      int khalf = (l >> 4) << 4;
      v16bf bv;
#pragma unroll
      for (int i = 0; i < 16; ++i)
        bv[i] = (__bf16)V[((size_t)b * NN + n0 + khalf + i) * DVC + col];
      sw_v[p] = bv;
    }
    __syncthreads();
    v16bf bf = sw_v[wave * 32 + lane];
#pragma unroll
    for (int mt = 0; mt < 16; ++mt)
      accs[mt] = __builtin_amdgcn_wmma_f32_16x16x32_bf16(
          false, sw_a[mt * 32 + lane], false, bf, (short)0, accs[mt], false, false);
    __syncthreads();
  }

  int half = lane >> 4, col = lane & 15;
#pragma unroll
  for (int mt = 0; mt < 16; ++mt)
#pragma unroll
    for (int r = 0; r < 8; ++r) {
      int m = mt * 16 + r + 8 * half;
      KVpart[(((size_t)b * 16 + chunk) * MMC + m) * DVC + wave * 16 + col] =
          accs[mt][r];
    }
}

// KV = sum over 16 chunks of KVpart (deterministic reduction)
__global__ __launch_bounds__(256) void kvred_kernel(const float* __restrict__ KVpart,
                                                    float* __restrict__ KV) {
  int idx = blockIdx.x * 256 + threadIdx.x;  // [0, BB*MMC*DVC)
  int b = idx >> 15;                          // MMC*DVC = 32768
  int off = idx & 32767;
  float s = 0.f;
#pragma unroll
  for (int c = 0; c < 16; ++c)
    s += KVpart[(((size_t)b * 16 + c) << 15) + off];
  KV[idx] = s;
}

// norm[b,n] = Qp[b,n,:] . Ksum[b,:] + eps  (one wave per row, coalesced)
__global__ __launch_bounds__(256) void norm_kernel(const __bf16* __restrict__ Qp,
                                                   const float* __restrict__ Ksum,
                                                   float* __restrict__ norm) {
  int wave = threadIdx.x >> 5, lane = threadIdx.x & 31;
  int gw = blockIdx.x * 8 + wave;  // [0, BB*NN)
  int b = gw >> 13;                // NN = 8192
  const __bf16* q = Qp + (size_t)gw * MMC;
  const float* ks = Ksum + b * MMC;
  float s = 0.f;
#pragma unroll
  for (int i = 0; i < MMC / 32; ++i) {
    int m = i * 32 + lane;
    s += (float)q[m] * ks[m];
  }
#pragma unroll
  for (int off = 16; off >= 1; off >>= 1) s += __shfl_xor(s, off, 32);
  if (lane == 0) norm[gw] = s + 1e-8f;
}

// out[b,n,v] = (Qp[b,n,:] @ KV[b,:,v]) / norm[b,n]
__global__ __launch_bounds__(256) void out_kernel(const __bf16* __restrict__ Qp,
                                                  const float* __restrict__ KV,
                                                  const float* __restrict__ norm,
                                                  float* __restrict__ out) {
  int wave = threadIdx.x >> 5, lane = threadIdx.x & 31;
  int gw = blockIdx.x * 8 + wave;  // 0..32767
  int vt = gw & 7;
  int nt = (gw >> 3) & (NN / 16 - 1);
  int b = gw >> 12;
  int row, kidx[16];
  a_layout(lane, kidx, row);
  int half = lane >> 4, col = lane & 15, khalf = half * 16;
  const __bf16* arow = Qp + ((size_t)b * NN + nt * 16 + row) * MMC;
  v8f acc = {0.f, 0.f, 0.f, 0.f, 0.f, 0.f, 0.f, 0.f};
#pragma unroll
  for (int ks = 0; ks < 8; ++ks) {
    AFragU af;
#pragma unroll
    for (int v = 0; v < 8; ++v)
      af.u[v] = *(const unsigned*)(arow + ks * 32 + kidx[2 * v]);
    v16bf bf;
#pragma unroll
    for (int i = 0; i < 16; ++i)
      bf[i] = (__bf16)KV[((size_t)b * MMC + ks * 32 + khalf + i) * DVC +
                         vt * 16 + col];
    acc = __builtin_amdgcn_wmma_f32_16x16x32_bf16(false, af.v, false, bf,
                                                  (short)0, acc, false, false);
  }
#pragma unroll
  for (int r = 0; r < 8; ++r) {
    int n = nt * 16 + r + 8 * half;
    out[((size_t)b * NN + n) * DVC + vt * 16 + col] =
        acc[r] / norm[(size_t)b * NN + n];
  }
}

extern "C" void kernel_launch(void* const* d_in, const int* in_sizes, int n_in,
                              void* d_out, int out_size, void* d_ws, size_t ws_size,
                              hipStream_t stream) {
  (void)in_sizes; (void)n_in; (void)out_size; (void)ws_size;
  const float* Q = (const float*)d_in[0];
  const float* K = (const float*)d_in[1];
  const float* V = (const float*)d_in[2];
  const float* mask = (const float*)d_in[3];
  const float* omega = (const float*)d_in[4];

  char* ws = (char*)d_ws;
  size_t off = 0;
  __bf16* Qp = (__bf16*)(ws + off);   off += (size_t)BB * NN * MMC * 2;   // 32 MB
  __bf16* KpT = (__bf16*)(ws + off);  off += (size_t)BB * MMC * NN * 2;   // 32 MB
  float* KVpart = (float*)(ws + off); off += (size_t)BB * 16 * MMC * DVC * 4;  // 16 MB
  float* KV = (float*)(ws + off);     off += (size_t)BB * MMC * DVC * 4;  // 1 MB
  float* Ksum = (float*)(ws + off);   off += (size_t)BB * MMC * 4;
  float* norm = (float*)(ws + off);   off += (size_t)BB * NN * 4;
  unsigned* maxK = (unsigned*)(ws + off); off += 64;
  float* out = (float*)d_out;

  const int smem_kmax = 64 * 32 * 32;              // 65536
  const int smem_phi = 64 * 32 * 32 + 256 * 4;     // 66560
  hipFuncSetAttribute((const void*)kmax_kernel,
                      hipFuncAttributeMaxDynamicSharedMemorySize, smem_kmax);
  hipFuncSetAttribute((const void*)phi_kernel<true>,
                      hipFuncAttributeMaxDynamicSharedMemorySize, smem_phi);
  hipFuncSetAttribute((const void*)phi_kernel<false>,
                      hipFuncAttributeMaxDynamicSharedMemorySize, smem_phi);

  init_kernel<<<1, 256, 0, stream>>>(maxK);
  kmax_kernel<<<BB * (NN / 128), 256, smem_kmax, stream>>>(K, omega, maxK);
  phi_kernel<false><<<BB * (NN / 128), 256, smem_phi, stream>>>(K, omega, mask,
                                                                maxK, KpT);
  phi_kernel<true><<<BB * (NN / 128), 256, smem_phi, stream>>>(Q, omega, mask,
                                                               maxK, Qp);
  ksum_kernel<<<BB * MMC / 8, 256, 0, stream>>>(KpT, Ksum);
  kv_kernel<<<BB * 16, 256, 0, stream>>>(KpT, V, KVpart);
  kvred_kernel<<<BB * MMC * DVC / 256, 256, 0, stream>>>(KVpart, KV);
  norm_kernel<<<BB * NN / 8, 256, 0, stream>>>(Qp, Ksum, norm);
  out_kernel<<<BB * (NN / 16), 256, 0, stream>>>(Qp, KV, norm, out);
}
